// GlobalModel_45492293599375
// MI455X (gfx1250) — matrix operand order repaired and checked by reading, not compiled, not run on hardware
//
#include <hip/hip_runtime.h>
#include <math.h>

typedef __attribute__((ext_vector_type(2)))  float    v2f;
typedef __attribute__((ext_vector_type(8)))  float    v8f;
typedef __attribute__((ext_vector_type(16))) _Float16 v16h;

#define B_GRAPHS 512
#define XIND     128
#define UIND     64
#define CIN      448   // UIND + 3*XIND
#define HS       512
#define SPLIT    8
#define EPS      1e-5f

__device__ __forceinline__ int lower_bound_i32(const int* __restrict__ a, int n, int v) {
    int lo = 0, hi = n;
    while (lo < hi) {
        int mid = (lo + hi) >> 1;
        if (a[mid] < v) lo = mid + 1; else hi = mid;
    }
    return lo;
}

// ---------------------------------------------------------------------------
// Phase 1: per-(segment, split) partial sum/max over contiguous node rows.
// Block = 128 threads = 4 waves; each wave streams one 512B row per iteration
// (lane q loads float4 at columns 4q..4q+3) -> fully coalesced HBM traffic.
// ---------------------------------------------------------------------------
__global__ void seg_partial_kernel(const float* __restrict__ x,
                                   const int* __restrict__ batch, int N,
                                   float* __restrict__ psum,
                                   float* __restrict__ pmax) {
    const int b   = blockIdx.x;
    const int s   = blockIdx.y;
    const int tid = threadIdx.x;
    const int q   = tid & 31;   // lane within wave -> 4 feature columns
    const int r   = tid >> 5;   // wave id -> row within group of 4

    const int start = lower_bound_i32(batch, N, b);
    const int end   = lower_bound_i32(batch, N, b + 1);

    float4 vs = make_float4(0.f, 0.f, 0.f, 0.f);
    float4 vm = make_float4(-INFINITY, -INFINITY, -INFINITY, -INFINITY);

    for (int row = start + s * 4 + r; row < end; row += SPLIT * 4) {
        float4 v = *(const float4*)(x + (size_t)row * XIND + q * 4);
        vs.x += v.x; vs.y += v.y; vs.z += v.z; vs.w += v.w;
        vm.x = fmaxf(vm.x, v.x); vm.y = fmaxf(vm.y, v.y);
        vm.z = fmaxf(vm.z, v.z); vm.w = fmaxf(vm.w, v.w);
    }

    __shared__ float ls[4 * XIND];
    __shared__ float lm[4 * XIND];
    const int cb = r * XIND + q * 4;
    ls[cb + 0] = vs.x; ls[cb + 1] = vs.y; ls[cb + 2] = vs.z; ls[cb + 3] = vs.w;
    lm[cb + 0] = vm.x; lm[cb + 1] = vm.y; lm[cb + 2] = vm.z; lm[cb + 3] = vm.w;
    __syncthreads();

    const int c = tid;  // one column per thread (128 columns)
    float ss = ls[c] + ls[XIND + c] + ls[2 * XIND + c] + ls[3 * XIND + c];
    float sm = fmaxf(fmaxf(lm[c], lm[XIND + c]),
                     fmaxf(lm[2 * XIND + c], lm[3 * XIND + c]));
    size_t o = ((size_t)s * B_GRAPHS + b) * XIND + c;
    psum[o] = ss;
    pmax[o] = sm;
}

// ---------------------------------------------------------------------------
// Phase 2: combine SPLIT partials, build feat = [u | smax | smean | ssum]
// ---------------------------------------------------------------------------
__global__ void feat_assemble_kernel(const float* __restrict__ u,
                                     const int* __restrict__ batch, int N,
                                     const float* __restrict__ psum,
                                     const float* __restrict__ pmax,
                                     float* __restrict__ feat) {
    const int b = blockIdx.x;
    const int c = threadIdx.x;  // 0..127
    float ss = 0.f, sm = -INFINITY;
#pragma unroll
    for (int s = 0; s < SPLIT; ++s) {
        size_t o = ((size_t)s * B_GRAPHS + b) * XIND + c;
        ss += psum[o];
        sm = fmaxf(sm, pmax[o]);
    }
    int cnt = lower_bound_i32(batch, N, b + 1) - lower_bound_i32(batch, N, b);
    float mean = ss / fmaxf((float)cnt, 1.0f);

    float* row = feat + (size_t)b * CIN;
    if (c < UIND) row[c] = u[b * UIND + c];
    row[UIND + c]            = sm;
    row[UIND + XIND + c]     = mean;
    row[UIND + 2 * XIND + c] = ss;
}

// ---------------------------------------------------------------------------
// WMMA GEMM: out[M=512, HS=512] = A[M,K] * W[HS,K]^T + bias
// One wave per 16x16 tile, V_WMMA_F32_16X16X4_F32 (full f32 precision).
// A frag: lane m=L%16, h=L/16 holds A[m0+m][k+2h+{0,1}]      (ISA 16x4 layout)
// B frag: lane holds W[n0+m][k+2h+{0,1}]                     (B rows across lanes)
// D frag: VGPR v -> out row m0+v+8h, col n0+m                (ISA 16x16 C/D layout)
// ---------------------------------------------------------------------------
__global__ void gemm_bias_wmma(const float* __restrict__ A,
                               const float* __restrict__ W,
                               const float* __restrict__ bias,
                               float* __restrict__ out, int K) {
    const int lane = threadIdx.x;
    const int m  = lane & 15;
    const int hh = lane >> 4;
    const int m0 = blockIdx.y << 4;
    const int n0 = blockIdx.x << 4;

    v8f acc = {0.f, 0.f, 0.f, 0.f, 0.f, 0.f, 0.f, 0.f};

#if __has_builtin(__builtin_amdgcn_wmma_f32_16x16x4_f32)
    const float* arow = A + (size_t)(m0 + m) * K + 2 * hh;
    const float* brow = W + (size_t)(n0 + m) * K + 2 * hh;
    for (int k = 0; k < K; k += 4) {
        v2f a = *(const v2f*)(arow + k);
        v2f b = *(const v2f*)(brow + k);
        acc = __builtin_amdgcn_wmma_f32_16x16x4_f32(
            false, a, false, b, (short)0, acc, false, false);
    }
#else
    // Fallback: f16 inputs, K-step 32 (codegen-confirmed builtin).
    const float* arow = A + (size_t)(m0 + m) * K;
    const float* brow = W + (size_t)(n0 + m) * K;
    for (int k = 0; k < K; k += 32) {
        v16h af, bf;
#pragma unroll
        for (int i = 0; i < 16; ++i) {
            int v = i >> 1, j = i & 1;
            int ka = 2 * v + j + 16 * hh;
            int kb = 2 * v + j + 16 * hh;
            af[i] = (_Float16)arow[k + ka];
            bf[i] = (_Float16)brow[k + kb];
        }
        acc = __builtin_amdgcn_wmma_f32_16x16x32_f16(
            false, af, false, bf, (short)0, acc, false, false);
    }
#endif

    const float bv = bias[n0 + m];
#pragma unroll
    for (int v = 0; v < 8; ++v) {
        int orow = m0 + v + 8 * hh;
        out[(size_t)orow * HS + n0 + m] = acc[v] + bv;
    }
}

// ---------------------------------------------------------------------------
// BatchNorm over batch dim: per-column mean and rsqrt(biased var + eps).
// One thread per column; consecutive lanes -> consecutive columns (coalesced).
// ---------------------------------------------------------------------------
__global__ void bn_stats_kernel(const float* __restrict__ hmat,
                                float* __restrict__ mean,
                                float* __restrict__ rsig) {
    const int col = blockIdx.x * blockDim.x + threadIdx.x;  // 512 total
    float s = 0.f, s2 = 0.f;
    for (int b = 0; b < B_GRAPHS; ++b) {
        float v = hmat[(size_t)b * HS + col];
        s += v; s2 += v * v;
    }
    float mu  = s  * (1.0f / B_GRAPHS);
    float var = s2 * (1.0f / B_GRAPHS) - mu * mu;
    mean[col] = mu;
    rsig[col] = rsqrtf(var + EPS);
}

__global__ void bn_apply_kernel(const float* __restrict__ src,
                                const float* __restrict__ mean,
                                const float* __restrict__ rsig,
                                const float* __restrict__ gamma,
                                const float* __restrict__ beta,
                                float* __restrict__ dst, int relu) {
    const int i = blockIdx.x * blockDim.x + threadIdx.x;
    if (i < B_GRAPHS * HS) {
        int col = i & (HS - 1);
        float v = (src[i] - mean[col]) * rsig[col] * gamma[col] + beta[col];
        if (relu) v = fmaxf(v, 0.f);
        dst[i] = v;
    }
}

extern "C" void kernel_launch(void* const* d_in, const int* in_sizes, int n_in,
                              void* d_out, int out_size, void* d_ws, size_t ws_size,
                              hipStream_t stream) {
    const float* x     = (const float*)d_in[0];
    const float* u     = (const float*)d_in[1];
    const int*   batch = (const int*)d_in[2];
    const float* W1    = (const float*)d_in[3];
    const float* b1    = (const float*)d_in[4];
    const float* g1v   = (const float*)d_in[5];
    const float* be1   = (const float*)d_in[6];
    const float* W2    = (const float*)d_in[7];
    const float* b2    = (const float*)d_in[8];
    const float* g2v   = (const float*)d_in[9];
    const float* be2   = (const float*)d_in[10];
    float* out = (float*)d_out;
    const int N = in_sizes[0] / XIND;

    float* ws   = (float*)d_ws;
    float* psum = ws;                                       // SPLIT*B*128 = 524288
    float* pmax = ws + (size_t)SPLIT * B_GRAPHS * XIND;     // 524288
    float* feat = pmax + (size_t)SPLIT * B_GRAPHS * XIND;   // 229376
    float* mean = feat + (size_t)B_GRAPHS * CIN;            // 512
    float* rsig = mean + HS;                                // 512
    float* hpre = psum;   // reused after feat assembly (262144 floats)
    float* h1   = pmax;   // reused after feat assembly (262144 floats)

    // Phase 1/2: segment reductions over x (the 512 MB, bandwidth-bound pass)
    seg_partial_kernel<<<dim3(B_GRAPHS, SPLIT), 128, 0, stream>>>(x, batch, N, psum, pmax);
    feat_assemble_kernel<<<B_GRAPHS, 128, 0, stream>>>(u, batch, N, psum, pmax, feat);

    // MLP layer 1: GEMM + bias, BN stats, BN+ReLU
    gemm_bias_wmma<<<dim3(HS / 16, B_GRAPHS / 16), 32, 0, stream>>>(feat, W1, b1, hpre, CIN);
    bn_stats_kernel<<<HS / 128, 128, 0, stream>>>(hpre, mean, rsig);
    bn_apply_kernel<<<(B_GRAPHS * HS) / 256, 256, 0, stream>>>(hpre, mean, rsig, g1v, be1, h1, 1);

    // MLP layer 2: GEMM + bias, BN stats, BN (no ReLU) -> d_out
    gemm_bias_wmma<<<dim3(HS / 16, B_GRAPHS / 16), 32, 0, stream>>>(h1, W2, b2, hpre, HS);
    bn_stats_kernel<<<HS / 128, 128, 0, stream>>>(hpre, mean, rsig);
    bn_apply_kernel<<<(B_GRAPHS * HS) / 256, 256, 0, stream>>>(hpre, mean, rsig, g2v, be2, out, 0);
}
